// AutoQVLALinearW4A16_56581899157845
// MI455X (gfx1250) — matrix operand order repaired and checked by reading, not compiled, not run on hardware
//
#include <hip/hip_runtime.h>
#include <hip/hip_bf16.h>

typedef __attribute__((ext_vector_type(16))) _Float16 v16h;
typedef __attribute__((ext_vector_type(8)))  _Float16 v8h;
typedef __attribute__((ext_vector_type(8)))  float    v8f;
typedef __attribute__((ext_vector_type(4)))  float    v4f;
typedef __attribute__((ext_vector_type(4)))  int      v4i;

#define M_DIM 8192
#define K_DIM 4096
#define N_DIM 11008
#define BM 128
#define BN 128
#define BK 64
#define LDA 72   // padded LDS row pitch in halves (144B -> conflict-free 16-row strided b128 reads)
#define LDB 72

__global__ __launch_bounds__(256)
void w4a16_wmma_gemm(const float* __restrict__ x,
                     const int*   __restrict__ qw,
                     const float* __restrict__ scales,
                     const float* __restrict__ bias,
                     float*       __restrict__ out)
{
    __shared__ _Float16 Ah[BM][LDA];   // x high f16 part
    __shared__ _Float16 Al[BM][LDA];   // x residual f16 part
    __shared__ _Float16 Bs[BN][LDB];   // dequantized int4 weights as f16 (scale folded into epilogue)

    const int tid  = threadIdx.x;
    const int lane = tid & 31;
    const int wave = tid >> 5;          // 0..7
    const int wm   = wave >> 2;         // 0..1  (64-row wave tile)
    const int wn   = wave & 3;          // 0..3  (32-col wave tile)
    const int m0   = blockIdx.y * BM;
    const int n0   = blockIdx.x * BN;

    // loader thread assignments
    const int ar = tid >> 1;            // A row 0..127
    const int ah = (tid & 1) * 32;      // A col offset within BK (32 floats each)
    const int br = tid & 127;           // B row 0..127
    const int bh = (tid >> 7) * 16;     // B packed-int32 offset (16 int32 = 32 K)

    const int l16 = lane & 15;
    const int lhi = lane >> 4;

    v8f acc[4][2] = {};

    for (int kb = 0; kb < K_DIM; kb += BK) {
        // ---------------- stage A: fp32 -> (hi,lo) f16 into LDS ----------------
        {
            const float* src = x + (size_t)(m0 + ar) * K_DIM + kb + ah;
            #pragma unroll
            for (int u = 0; u < 4; ++u) {
                v4f f0 = *(const v4f*)(src + u * 8);
                v4f f1 = *(const v4f*)(src + u * 8 + 4);
                float fl[8] = {f0.x, f0.y, f0.z, f0.w, f1.x, f1.y, f1.z, f1.w};
                v8h hh, ll;
                #pragma unroll
                for (int e = 0; e < 8; ++e) {
                    _Float16 h = (_Float16)fl[e];
                    hh[e] = h;
                    ll[e] = (_Float16)(fl[e] - (float)h);   // residual: two-term f16 split
                }
                *(v8h*)&Ah[ar][ah + u * 8] = hh;
                *(v8h*)&Al[ar][ah + u * 8] = ll;
            }
        }
        // ---------------- stage B: packed int4 -> signed f16 into LDS ----------------
        {
            const int* srcq = qw + (size_t)(n0 + br) * (K_DIM / 2) + (kb >> 1) + bh;
            #pragma unroll
            for (int u = 0; u < 4; ++u) {
                v4i q = *(const v4i*)(srcq + u * 4);
                int qa[4] = {q.x, q.y, q.z, q.w};
                v8h wv;
                #pragma unroll
                for (int e = 0; e < 4; ++e) {
                    int b   = qa[e] & 0xff;
                    int lo  = b & 15;           // low nibble first (matches reference)
                    int hi  = (b >> 4) & 15;
                    lo -= (lo > 7) ? 16 : 0;    // map [8,15] -> [-8,-1]
                    hi -= (hi > 7) ? 16 : 0;
                    wv[2 * e]     = (_Float16)lo;
                    wv[2 * e + 1] = (_Float16)hi;
                }
                *(v8h*)&Bs[br][2 * bh + u * 8] = wv;
            }
        }

        // prefetch next K-stage while this one is consumed
        if (kb + BK < K_DIM) {
            __builtin_prefetch(x  + (size_t)(m0 + ar) * K_DIM + kb + BK + ah, 0, 0);
            __builtin_prefetch(qw + (size_t)(n0 + br) * (K_DIM / 2) + ((kb + BK) >> 1) + bh, 0, 0);
        }

        __syncthreads();

        // ---------------- compute: 2 K-steps of 32, 16 WMMAs each ----------------
        #pragma unroll
        for (int ks = 0; ks < BK; ks += 32) {
            // B operands: lane<16 -> column N=l16, K = ks+0..15 contiguous; lane>=16 -> K += 16
            v16h breg[2];
            #pragma unroll
            for (int j = 0; j < 2; ++j) {
                const _Float16* bp = &Bs[wn * 32 + j * 16 + l16][ks + lhi * 16];
                union { v16h v; v8h h[2]; } ub;
                ub.h[0] = *(const v8h*)(bp);
                ub.h[1] = *(const v8h*)(bp + 8);
                breg[j] = ub.v;
            }
            #pragma unroll
            for (int i = 0; i < 4; ++i) {
                // A operand: row M = l16, K = ko+0..15 contiguous (ko = ks + 16*(lane>>4))
                const int row = wm * 64 + i * 16 + l16;
                const int ko  = ks + lhi * 16;
                union { v16h v; v8h h[2]; } ua, ul;
                ua.h[0] = *(const v8h*)&Ah[row][ko];
                ua.h[1] = *(const v8h*)&Ah[row][ko + 8];
                ul.h[0] = *(const v8h*)&Al[row][ko];
                ul.h[1] = *(const v8h*)&Al[row][ko + 8];
                #pragma unroll
                for (int j = 0; j < 2; ++j) {
                    acc[i][j] = __builtin_amdgcn_wmma_f32_16x16x32_f16(
                        false, ua.v, false, breg[j], (short)0, acc[i][j], false, false);
                    acc[i][j] = __builtin_amdgcn_wmma_f32_16x16x32_f16(
                        false, ul.v, false, breg[j], (short)0, acc[i][j], false, false);
                }
            }
        }
        __syncthreads();
    }

    // ---------------- epilogue: y = acc * scale[n] + bias[n] ----------------
    #pragma unroll
    for (int j = 0; j < 2; ++j) {
        const int n = n0 + wn * 32 + j * 16 + l16;
        const float s  = scales[n];
        const float bb = bias[n];
        #pragma unroll
        for (int i = 0; i < 4; ++i) {
            const int mbase = m0 + wm * 64 + i * 16 + lhi * 8;
            #pragma unroll
            for (int r = 0; r < 8; ++r) {
                out[(size_t)(mbase + r) * N_DIM + n] = acc[i][j][r] * s + bb;
            }
        }
    }
}

extern "C" void kernel_launch(void* const* d_in, const int* in_sizes, int n_in,
                              void* d_out, int out_size, void* d_ws, size_t ws_size,
                              hipStream_t stream) {
    const float* xin   = (const float*)d_in[0];
    const int*   qw    = (const int*)d_in[1];
    const float* sc    = (const float*)d_in[2];
    const float* bi    = (const float*)d_in[3];
    float*       out   = (float*)d_out;

    dim3 grid(N_DIM / BN, M_DIM / BM);   // 86 x 64
    w4a16_wmma_gemm<<<grid, dim3(256), 0, stream>>>(xin, qw, sc, bi, out);
}
